// GCN_39350490366357
// MI455X (gfx1250) — matrix operand order repaired and checked by reading, not compiled, not run on hardware
//
#include <hip/hip_runtime.h>
#include <hip/hip_bf16.h>

typedef __attribute__((ext_vector_type(16))) __bf16 v16bf;
typedef __attribute__((ext_vector_type(16))) float  v16f;
typedef __attribute__((ext_vector_type(8)))  float  v8f;

#define D_FEAT 512
#define HIDDEN 128
#define NCLS   40
#define NCLSP  48

__device__ __forceinline__ unsigned short f2bf(float f) {
    unsigned u = __float_as_uint(f);
    return (unsigned short)((u + 0x7FFFu + ((u >> 16) & 1u)) >> 16);
}

// ---------------- utility kernels ----------------

__global__ void gcn_zero_f4(float4* __restrict__ p, int n4) {
    int i = blockIdx.x * blockDim.x + threadIdx.x;
    if (i < n4) p[i] = make_float4(0.f, 0.f, 0.f, 0.f);
}

__global__ void gcn_init_deg(float* __restrict__ deg, int n) {
    int i = blockIdx.x * blockDim.x + threadIdx.x;
    if (i < n) deg[i] = 1.0f;          // self-loop contributes 1
}

__global__ void gcn_deg_accum(const int* __restrict__ dst, int E, float* __restrict__ deg) {
    int i = blockIdx.x * blockDim.x + threadIdx.x;
    if (i < E) atomicAdd(&deg[dst[i]], 1.0f);
}

__global__ void gcn_dinv(float* __restrict__ deg, int n) {
    int i = blockIdx.x * blockDim.x + threadIdx.x;
    if (i < n) {
        float d = deg[i];
        deg[i] = (d > 0.f) ? rsqrtf(d) : 0.f;   // dinv stored in place
    }
}

// Wt[n*K + k] = bf16(W[k*Nsrc + n]) for n < Nsrc, else 0  (transpose + convert + pad)
__global__ void gcn_prep_wt(const float* __restrict__ W, unsigned short* __restrict__ Wt,
                            int K, int Nsrc, int Npad) {
    int i = blockIdx.x * blockDim.x + threadIdx.x;
    int total = Npad * K;
    if (i < total) {
        int n = i / K, k = i - n * K;
        Wt[i] = (n < Nsrc) ? f2bf(W[(size_t)k * Nsrc + n]) : (unsigned short)0;
    }
}

// ---------------- WMMA GEMM ----------------
// out[M x ldo] = A[M x K](f32) * Wt[NT*16 x K](bf16)^T
// One wave owns one 16-row strip and ALL NT column tiles: A is loaded and
// converted (v_cvt_pk_bf16_f32 via convertvector) once per K-step, then NT
// independent back-to-back WMMAs accumulate into NT register tiles.
// K multiple of 32; M multiple of 16.

template <int NT>
__global__ void __launch_bounds__(256)
gcn_gemm_bf16_wmma(const float* __restrict__ A, const __bf16* __restrict__ Wt,
                   float* __restrict__ out, int K, int ldo, int mtiles) {
    int wave = (blockIdx.x * blockDim.x + threadIdx.x) >> 5;
    int lane = threadIdx.x & 31;
    if (wave >= mtiles) return;                 // wave-uniform: EXEC stays all-1
    int l15 = lane & 15;
    int hi  = lane >> 4;                        // lane half selects K-subblock

    const float*  arow  = A  + (size_t)(wave * 16 + l15) * (size_t)K;
    const __bf16* wbase = Wt + (size_t)l15 * (size_t)K;

    v8f acc[NT];
    #pragma unroll
    for (int t = 0; t < NT; ++t) acc[t] = (v8f){};

    int ksteps = K >> 5;
    for (int kk = 0; kk < ksteps; ++kk) {
        // A (16x32 bf16): lane<16 -> K {0..7,16..23}; lane>=16 -> K {8..15,24..31}
        int ka = (kk << 5) + hi * 8;
        float4 f0 = *(const float4*)(arow + ka);
        float4 f1 = *(const float4*)(arow + ka + 4);
        float4 f2 = *(const float4*)(arow + ka + 16);
        float4 f3 = *(const float4*)(arow + ka + 20);
        v16f af = {f0.x, f0.y, f0.z, f0.w, f1.x, f1.y, f1.z, f1.w,
                   f2.x, f2.y, f2.z, f2.w, f3.x, f3.y, f3.z, f3.w};
        v16bf a = __builtin_convertvector(af, v16bf);   // packed v_cvt_pk_bf16_f32

        #pragma unroll
        for (int t = 0; t < NT; ++t) {
            // B (32x16 bf16): lane<16 -> K {0..7,16..23} of column l15; lane>=16 -> K {8..15,24..31}
            const uint4* bp = (const uint4*)(wbase + (size_t)t * 16 * K + (kk << 5) + hi * 8);
            union { uint4 q[2]; v16bf v; } b;
            b.q[0] = bp[0];
            b.q[1] = bp[2];
            acc[t] = __builtin_amdgcn_wmma_f32_16x16x32_bf16(
                /*neg_a=*/false, a, /*neg_b=*/false, b.v,
                /*c_mod=*/(short)0, acc[t], /*reuse_a=*/false, /*reuse_b=*/false);
        }
    }

    // C/D layout: VGPR r -> row M = r (lanes 0-15) or r+8 (lanes 16-31); col = lane&15
    int rbase = wave * 16 + hi * 8;
    #pragma unroll
    for (int t = 0; t < NT; ++t) {
        int cc = t * 16 + l15;
        #pragma unroll
        for (int r = 0; r < 8; ++r)
            out[(size_t)(rbase + r) * (size_t)ldo + cc] = acc[t][r];
    }
}

// ---------------- edge aggregation ----------------

// layer1: 32 lanes per edge, float4 per lane over 128 feats
__global__ void gcn_scatter128(const int* __restrict__ src, const int* __restrict__ dst,
                               const float* __restrict__ dinv, const float* __restrict__ h,
                               float* __restrict__ agg, int E) {
    int tid = blockIdx.x * blockDim.x + threadIdx.x;
    int e = tid >> 5;
    if (e >= E) return;
    int c4 = (tid & 31) << 2;
    int s = src[e], d = dst[e];
    float nrm = dinv[s] * dinv[d];
    float4 v = *(const float4*)(h + (size_t)s * HIDDEN + c4);
    float* ag = agg + (size_t)d * HIDDEN + c4;
    atomicAdd(ag + 0, v.x * nrm);
    atomicAdd(ag + 1, v.y * nrm);
    atomicAdd(ag + 2, v.z * nrm);
    atomicAdd(ag + 3, v.w * nrm);
}

// layer2: one thread per (edge, class); h2 stride = NCLSP (padded)
__global__ void gcn_scatter40(const int* __restrict__ src, const int* __restrict__ dst,
                              const float* __restrict__ dinv, const float* __restrict__ h2,
                              float* __restrict__ agg, int E) {
    int tid = blockIdx.x * blockDim.x + threadIdx.x;
    int e = tid / NCLS;
    if (e >= E) return;
    int c = tid - e * NCLS;
    int s = src[e], d = dst[e];
    float nrm = dinv[s] * dinv[d];
    float v = h2[(size_t)s * NCLSP + c];
    atomicAdd(&agg[(size_t)d * NCLS + c], v * nrm);
}

// ---------------- epilogues ----------------

// agg1 = relu(agg1 + h1 * dinv^2 + b1)   (in place; becomes layer-2 input)
__global__ void gcn_finalize_relu(float* __restrict__ agg, const float* __restrict__ h1,
                                  const float* __restrict__ dinv, const float* __restrict__ b1,
                                  int n) {
    int tid = blockIdx.x * blockDim.x + threadIdx.x;
    int node = tid >> 5;
    if (node >= n) return;
    int c4 = (tid & 31) << 2;
    float di = dinv[node];
    float sl = di * di;
    float4 a  = *(const float4*)(agg + (size_t)node * HIDDEN + c4);
    float4 hh = *(const float4*)(h1  + (size_t)node * HIDDEN + c4);
    float4 bb = *(const float4*)(b1 + c4);
    float4 r;
    r.x = fmaxf(a.x + hh.x * sl + bb.x, 0.f);
    r.y = fmaxf(a.y + hh.y * sl + bb.y, 0.f);
    r.z = fmaxf(a.z + hh.z * sl + bb.z, 0.f);
    r.w = fmaxf(a.w + hh.w * sl + bb.w, 0.f);
    *(float4*)(agg + (size_t)node * HIDDEN + c4) = r;
}

// out = log_softmax(agg2 + h2*dinv^2 + b2) per node
__global__ void gcn_finalize_lsm(const float* __restrict__ agg2, const float* __restrict__ h2,
                                 const float* __restrict__ dinv, const float* __restrict__ b2,
                                 float* __restrict__ out, int n) {
    int i = blockIdx.x * blockDim.x + threadIdx.x;
    if (i >= n) return;
    float di = dinv[i];
    float sl = di * di;
    float v[NCLS];
    float mx = -1e30f;
    #pragma unroll
    for (int c = 0; c < NCLS; ++c) {
        float t = agg2[(size_t)i * NCLS + c] + h2[(size_t)i * NCLSP + c] * sl + b2[c];
        v[c] = t;
        mx = fmaxf(mx, t);
    }
    float s = 0.f;
    #pragma unroll
    for (int c = 0; c < NCLS; ++c) s += expf(v[c] - mx);
    float ls = logf(s);
    #pragma unroll
    for (int c = 0; c < NCLS; ++c)
        out[(size_t)i * NCLS + c] = v[c] - mx - ls;
}

// ---------------- launcher ----------------

extern "C" void kernel_launch(void* const* d_in, const int* in_sizes, int n_in,
                              void* d_out, int out_size, void* d_ws, size_t ws_size,
                              hipStream_t stream) {
    const float* x  = (const float*)d_in[0];
    const int*   ei = (const int*)d_in[1];   // [2, E] int32
    const float* W1 = (const float*)d_in[2];
    const float* b1 = (const float*)d_in[3];
    const float* W2 = (const float*)d_in[4];
    const float* b2 = (const float*)d_in[5];
    float* out = (float*)d_out;

    int n = in_sizes[0] / D_FEAT;   // 100000
    int E = in_sizes[1] / 2;        // 1600000
    const int* src = ei;
    const int* dst = ei + E;

    // workspace layout
    char* base = (char*)d_ws;
    size_t off = 0;
    float* deg = (float*)(base + off);                 // n floats -> becomes dinv
    off += ((size_t)n * 4 + 511) & ~(size_t)511;
    float* h1 = (float*)(base + off);                  // n*128 f32 (reused as h2raw, stride 48)
    off += (size_t)n * HIDDEN * 4;
    float* agg1 = (float*)(base + off);                // n*128 f32 (in-place relu -> layer2 input)
    off += (size_t)n * HIDDEN * 4;
    float* agg2 = (float*)(base + off);                // n*40 f32
    off += (size_t)n * NCLS * 4;
    unsigned short* Wt1 = (unsigned short*)(base + off);  // 128*512 bf16
    off += (size_t)HIDDEN * D_FEAT * 2;
    unsigned short* Wt2 = (unsigned short*)(base + off);  // 48*128 bf16 (zero-padded)
    float* h2raw = h1;                                  // alias: h1 dead after finalize_relu

    const int T = 256;
    // zero accumulators (must re-zero every call)
    gcn_zero_f4<<<((size_t)n * (HIDDEN / 4) + T - 1) / T, T, 0, stream>>>((float4*)agg1, n * (HIDDEN / 4));
    gcn_zero_f4<<<((size_t)n * (NCLS / 4) + T - 1) / T, T, 0, stream>>>((float4*)agg2, n * (NCLS / 4));
    // degree + norm
    gcn_init_deg<<<(n + T - 1) / T, T, 0, stream>>>(deg, n);
    gcn_deg_accum<<<(E + T - 1) / T, T, 0, stream>>>(dst, E, deg);
    gcn_dinv<<<(n + T - 1) / T, T, 0, stream>>>(deg, n);
    // weight transpose + bf16 convert
    gcn_prep_wt<<<(HIDDEN * D_FEAT + T - 1) / T, T, 0, stream>>>(W1, Wt1, D_FEAT, HIDDEN, HIDDEN);
    gcn_prep_wt<<<(NCLSP * HIDDEN + T - 1) / T, T, 0, stream>>>(W2, Wt2, HIDDEN, NCLS, NCLSP);

    // layer 1: h1 = x @ W1  (WMMA bf16, f32 accum; 8 N-tiles per wave)
    int mtiles = n / 16;                    // 6250 exactly
    gcn_gemm_bf16_wmma<HIDDEN / 16><<<((size_t)mtiles * 32 + T - 1) / T, T, 0, stream>>>(
        x, (const __bf16*)Wt1, h1, D_FEAT, HIDDEN, mtiles);
    gcn_scatter128<<<((size_t)E * 32 + T - 1) / T, T, 0, stream>>>(src, dst, deg, h1, agg1, E);
    gcn_finalize_relu<<<((size_t)n * 32 + T - 1) / T, T, 0, stream>>>(agg1, h1, deg, b1, n);

    // layer 2: h2raw = relu(h1_agg) @ W2 (N padded to 48; 3 N-tiles per wave)
    gcn_gemm_bf16_wmma<NCLSP / 16><<<((size_t)mtiles * 32 + T - 1) / T, T, 0, stream>>>(
        agg1, (const __bf16*)Wt2, h2raw, HIDDEN, NCLSP, mtiles);
    gcn_scatter40<<<((size_t)E * NCLS + T - 1) / T, T, 0, stream>>>(src, dst, deg, h2raw, agg2, E);
    gcn_finalize_lsm<<<(n + T - 1) / T, T, 0, stream>>>(agg2, h2raw, deg, b2, out, n);

    (void)n_in; (void)out_size; (void)ws_size;
}
